// MultiHeadAdditiveAttentionLayer_2473901163182
// MI455X (gfx1250) — compile-verified, matrix operand order checked by reading, not run on hardware
//
#include <hip/hip_runtime.h>
#include <hip/hip_bf16.h>

// ---------------------------------------------------------------------------
// MultiHeadAdditiveAttention for MI455X (gfx1250, wave32, WMMA)
// B=2, L=384, HID=512, HEADS=8, HD=64
// ---------------------------------------------------------------------------

#define B_   2
#define L_   384
#define HID_ 512
#define H_   8
#define HD_  64
#define R_   (B_ * L_)          // 768 rows of the merged (b,l) dimension

typedef __attribute__((ext_vector_type(16))) _Float16 v16h;
typedef __attribute__((ext_vector_type(8)))  _Float16 v8h;
typedef __attribute__((ext_vector_type(8)))  float    v8f;

// ------------------------------ fast tanh ----------------------------------
__device__ __forceinline__ float fast_tanh(float x) {
#if __has_builtin(__builtin_amdgcn_tanhf)
    return __builtin_amdgcn_tanhf(x);          // gfx1250 v_tanh_f32
#else
    // tanh(x) = 1 - 2/(exp2(2*log2e*x)+1)
    float e = __builtin_amdgcn_exp2f(x * 2.8853900817779268f);
    return 1.0f - 2.0f * __builtin_amdgcn_rcpf(e + 1.0f);
#endif
}

// -------------------------- conversion kernels -----------------------------
__global__ void cvt_f32_to_f16(const float* __restrict__ src,
                               _Float16* __restrict__ dst, int n) {
    int i = blockIdx.x * blockDim.x + threadIdx.x;
    if (i < n) dst[i] = (_Float16)src[i];
}

// src is (rows=K, cols=N) row-major; dst is BT layout (N x K) row-major.
__global__ void cvt_transpose_f32_to_f16(const float* __restrict__ src,
                                         _Float16* __restrict__ dst,
                                         int rows, int cols) {
    int i = blockIdx.x * blockDim.x + threadIdx.x;
    if (i < rows * cols) {
        int r = i / cols, c = i % cols;
        dst[(long)c * rows + r] = (_Float16)src[i];
    }
}

// ------------------------- generic batched WMMA GEMM -----------------------
// C[M,N] = A[M,K] @ B[K,N] (+bias[n]); B supplied pre-transposed as BT[N,K].
// Batch index z = b*H + h with independent (b,h) strides on A/BT/C.
// One wave computes a 32x32 tile: 2 A-frags x 2 B-frags -> 4 accumulators,
// i.e. 2 global_load_b128 per v_wmma (vs 4 for a 16x16 tile).
// M and N must be multiples of 32, K a multiple of 32.
template <typename OutT, bool TSTORE>
__global__ void wmma_gemm_kernel(const _Float16* __restrict__ A, int lda,
                                 long sAb, long sAh,
                                 const _Float16* __restrict__ BT, int ldb,
                                 long sBb, long sBh,
                                 const float* __restrict__ bias,
                                 OutT* __restrict__ C, int ldc,
                                 long sCb, long sCh,
                                 int M, int N, int K, int H) {
    int z  = blockIdx.z;
    int bb = z / H, hh = z % H;
    A  += bb * sAb + hh * sAh;
    BT += bb * sBb + hh * sBh;
    C  += bb * sCb + hh * sCh;

    int wave   = (blockIdx.x * blockDim.x + threadIdx.x) >> 5;
    int tilesN = N >> 5;
    int tiles  = (M >> 5) * tilesN;
    if (wave >= tiles) return;                 // wave-uniform; EXEC stays full
    int tm = wave / tilesN, tn = wave % tilesN;

    int lane = threadIdx.x & 31;
    int half = lane >> 4;                      // selects K-octet
    int lmod = lane & 15;                      // M (A) / N (BT) index

    const _Float16* arow0 = A  + (long)(tm * 32 + lmod) * lda;
    const _Float16* arow1 = arow0 + (long)16 * lda;
    const _Float16* brow0 = BT + (long)(tn * 32 + lmod) * ldb;
    const _Float16* brow1 = brow0 + (long)16 * ldb;

    v8f acc00 = {}, acc01 = {}, acc10 = {}, acc11 = {};
    for (int k0 = 0; k0 < K; k0 += 32) {
        int kb = k0 + half * 8;
        union { v16h v; v8h p[2]; } a0, a1, b0, b1;
        // 16-bit A fragment: idx0..7 -> K=kb..kb+7, idx8..15 -> K=kb+16..kb+23
        a0.p[0] = *reinterpret_cast<const v8h*>(arow0 + kb);
        a0.p[1] = *reinterpret_cast<const v8h*>(arow0 + kb + 16);
        a1.p[0] = *reinterpret_cast<const v8h*>(arow1 + kb);
        a1.p[1] = *reinterpret_cast<const v8h*>(arow1 + kb + 16);
        b0.p[0] = *reinterpret_cast<const v8h*>(brow0 + kb);
        b0.p[1] = *reinterpret_cast<const v8h*>(brow0 + kb + 16);
        b1.p[0] = *reinterpret_cast<const v8h*>(brow1 + kb);
        b1.p[1] = *reinterpret_cast<const v8h*>(brow1 + kb + 16);
        acc00 = __builtin_amdgcn_wmma_f32_16x16x32_f16(
                    false, a0.v, false, b0.v, (short)0, acc00, false, false);
        acc01 = __builtin_amdgcn_wmma_f32_16x16x32_f16(
                    false, a0.v, false, b1.v, (short)0, acc01, false, false);
        acc10 = __builtin_amdgcn_wmma_f32_16x16x32_f16(
                    false, a1.v, false, b0.v, (short)0, acc10, false, false);
        acc11 = __builtin_amdgcn_wmma_f32_16x16x32_f16(
                    false, a1.v, false, b1.v, (short)0, acc11, false, false);
    }

    int   n0 = tn * 32 + lmod;
    int   n1 = n0 + 16;
    float bv0 = bias ? bias[n0] : 0.0f;
    float bv1 = bias ? bias[n1] : 0.0f;

    auto store_tile = [&](const v8f& acc, int mbase, int n, float bval) {
#pragma unroll
        for (int j = 0; j < 8; ++j) {
            int   m = mbase + half * 8 + j;    // documented C/D striping
            float r = acc[j] + bval;
            if (TSTORE) C[(long)n * ldc + m] = (OutT)r;
            else        C[(long)m * ldc + n] = (OutT)r;
        }
    };
    int m0 = tm * 32, m1 = m0 + 16;
    store_tile(acc00, m0, n0, bv0);
    store_tile(acc01, m0, n1, bv1);
    store_tile(acc10, m1, n0, bv0);
    store_tile(acc11, m1, n1, bv1);
}

// ----------------------------- energy kernel -------------------------------
// energy[b,h,q,k] = (1/8) * sum_d tanh(qW[b,q,h*64+d] + kU[b,k,h*64+d]) * vv[d]
// 32q x 32k tile per block through LDS; 2x2 micro-tile per thread so each
// pair of LDS reads feeds 4 tanh evaluations (tanh is the FLOP bottleneck).
__global__ void energy_kernel(const float* __restrict__ qW,
                              const float* __restrict__ kU,
                              const float* __restrict__ vv,
                              float* __restrict__ energy) {
    __shared__ float sq[32][HD_ + 4];
    __shared__ float sk[32][HD_ + 4];
    __shared__ float sv[HD_];

    int z = blockIdx.z;              // b*8 + h
    int b = z >> 3, h = z & 7;
    int q0 = blockIdx.x * 32, k0 = blockIdx.y * 32;
    int tx = threadIdx.x, ty = threadIdx.y;
    int tid = ty * 16 + tx;

    const float* qbase = qW + ((long)(b * L_ + q0)) * HID_ + h * HD_;
    const float* kbase = kU + ((long)(b * L_ + k0)) * HID_ + h * HD_;
    for (int t = tid; t < 32 * HD_; t += 256) {
        int r = t >> 6, d = t & 63;
        sq[r][d] = qbase[(long)r * HID_ + d];
        sk[r][d] = kbase[(long)r * HID_ + d];
    }
    if (tid < HD_) sv[tid] = vv[tid];
    __syncthreads();

    float a00 = 0.0f, a01 = 0.0f, a10 = 0.0f, a11 = 0.0f;
#pragma unroll 4
    for (int d = 0; d < HD_; ++d) {
        float qa = sq[ty][d],      qb = sq[ty + 16][d];
        float ka = sk[tx][d],      kb = sk[tx + 16][d];
        float w  = sv[d];
        a00 += fast_tanh(qa + ka) * w;
        a01 += fast_tanh(qa + kb) * w;
        a10 += fast_tanh(qb + ka) * w;
        a11 += fast_tanh(qb + kb) * w;
    }

    float* erow0 = energy + (((long)z * L_ + q0 + ty) * L_) + k0;
    float* erow1 = erow0 + (long)16 * L_;
    erow0[tx]      = a00 * 0.125f;
    erow0[tx + 16] = a01 * 0.125f;
    erow1[tx]      = a10 * 0.125f;
    erow1[tx + 16] = a11 * 0.125f;
}

// ---------------------------- softmax kernel -------------------------------
// One wave32 per (b,h,q) row of 384; in-place f32 + f16 copy for attn@V WMMA.
__global__ void softmax_kernel(float* __restrict__ attn,
                               _Float16* __restrict__ attn16) {
    int wid  = (blockIdx.x * blockDim.x + threadIdx.x) >> 5;
    int lane = threadIdx.x & 31;
    float*    row   = attn   + (long)wid * L_;
    _Float16* row16 = attn16 + (long)wid * L_;

    float v[12];
    float mx = -3.4e38f;
#pragma unroll
    for (int i = 0; i < 12; ++i) {
        v[i] = row[lane + i * 32];
        mx = fmaxf(mx, v[i]);
    }
#pragma unroll
    for (int s = 16; s; s >>= 1) mx = fmaxf(mx, __shfl_xor(mx, s, 32));

    float sum = 0.0f;
#pragma unroll
    for (int i = 0; i < 12; ++i) {
        v[i] = __builtin_amdgcn_exp2f((v[i] - mx) * 1.4426950408889634f);
        sum += v[i];
    }
#pragma unroll
    for (int s = 16; s; s >>= 1) sum += __shfl_xor(sum, s, 32);

    float inv = 1.0f / sum;
#pragma unroll
    for (int i = 0; i < 12; ++i) {
        float p = v[i] * inv;
        row[lane + i * 32]   = p;
        row16[lane + i * 32] = (_Float16)p;
    }
}

// ------------------------------- launcher ----------------------------------
extern "C" void kernel_launch(void* const* d_in, const int* in_sizes, int n_in,
                              void* d_out, int out_size, void* d_ws, size_t ws_size,
                              hipStream_t stream) {
    const float* query = (const float*)d_in[0];
    const float* key   = (const float*)d_in[1];
    const float* value = (const float*)d_in[2];
    const float* Wq = (const float*)d_in[3];  const float* bq = (const float*)d_in[4];
    const float* Wk = (const float*)d_in[5];  const float* bk = (const float*)d_in[6];
    const float* Wv = (const float*)d_in[7];  const float* bv = (const float*)d_in[8];
    const float* Ww = (const float*)d_in[9];  const float* bw = (const float*)d_in[10];
    const float* Wu = (const float*)d_in[11]; const float* bu = (const float*)d_in[12];
    const float* vv = (const float*)d_in[13];
    const float* Wo = (const float*)d_in[14]; const float* bo = (const float*)d_in[15];

    float* xOut    = (float*)d_out;                 // [768, 512]
    float* attnOut = xOut + (long)R_ * HID_;        // [2,8,384,384] f32

    // ---- workspace carve (256B aligned) ----
    char* ws = (char*)d_ws;
    size_t off = 0;
    auto take = [&](size_t bytes) -> void* {
        void* p = ws + off;
        off = (off + bytes + 255) & ~(size_t)255;
        return p;
    };
    const size_t nAct = (size_t)R_ * HID_;          // 393216
    const size_t nAtt = (size_t)B_ * H_ * L_ * L_;  // 2359296

    _Float16* q16   = (_Float16*)take(nAct * 2);
    _Float16* k16   = (_Float16*)take(nAct * 2);
    _Float16* v16   = (_Float16*)take(nAct * 2);
    _Float16* WqT   = (_Float16*)take((size_t)HID_ * HID_ * 2);
    _Float16* WkT   = (_Float16*)take((size_t)HID_ * HID_ * 2);
    _Float16* WvT   = (_Float16*)take((size_t)HID_ * HID_ * 2);
    _Float16* WoT   = (_Float16*)take((size_t)HID_ * HID_ * 2);
    _Float16* WwT   = (_Float16*)take((size_t)HD_ * HD_ * 2);
    _Float16* WuT   = (_Float16*)take((size_t)HD_ * HD_ * 2);
    _Float16* qh    = (_Float16*)take(nAct * 2);    // Q projected, f16 [768,512]
    _Float16* kh    = (_Float16*)take(nAct * 2);    // K projected, f16 [768,512]
    _Float16* vhT   = (_Float16*)take(nAct * 2);    // V projected, f16 TRANSPOSED [512,768]
    float*    qW32  = (float*)   take(nAct * 4);    // Q@Ww+bw, f32 [768,512]
    float*    kU32  = (float*)   take(nAct * 4);    // K@Wu+bu, f32 [768,512]
    _Float16* attn16= (_Float16*)take(nAtt * 2);    // softmax probs, f16
    _Float16* avh   = (_Float16*)take(nAct * 2);    // attn@V merged heads, f16

    // ---- 1. convert activations & weights to f16 (weights transposed) ----
    cvt_f32_to_f16<<<(int)(nAct / 256), 256, 0, stream>>>(query, q16, (int)nAct);
    cvt_f32_to_f16<<<(int)(nAct / 256), 256, 0, stream>>>(key,   k16, (int)nAct);
    cvt_f32_to_f16<<<(int)(nAct / 256), 256, 0, stream>>>(value, v16, (int)nAct);
    cvt_transpose_f32_to_f16<<<HID_ * HID_ / 256, 256, 0, stream>>>(Wq, WqT, HID_, HID_);
    cvt_transpose_f32_to_f16<<<HID_ * HID_ / 256, 256, 0, stream>>>(Wk, WkT, HID_, HID_);
    cvt_transpose_f32_to_f16<<<HID_ * HID_ / 256, 256, 0, stream>>>(Wv, WvT, HID_, HID_);
    cvt_transpose_f32_to_f16<<<HID_ * HID_ / 256, 256, 0, stream>>>(Wo, WoT, HID_, HID_);
    cvt_transpose_f32_to_f16<<<HD_ * HD_ / 256, 256, 0, stream>>>(Ww, WwT, HD_, HD_);
    cvt_transpose_f32_to_f16<<<HD_ * HD_ / 256, 256, 0, stream>>>(Wu, WuT, HD_, HD_);

    // ---- 2. Q/K/V projections (WMMA f16 -> f16 out; V stored transposed) ----
    {   // 32x32 tiles: (768/32)*(512/32) = 384 tiles; 4 waves/block -> 96 blocks
        dim3 g(96, 1, 1), blk(128);
        wmma_gemm_kernel<_Float16, false><<<g, blk, 0, stream>>>(
            q16, HID_, 0, 0, WqT, HID_, 0, 0, bq, qh, HID_, 0, 0, R_, HID_, HID_, 1);
        wmma_gemm_kernel<_Float16, false><<<g, blk, 0, stream>>>(
            k16, HID_, 0, 0, WkT, HID_, 0, 0, bk, kh, HID_, 0, 0, R_, HID_, HID_, 1);
        wmma_gemm_kernel<_Float16, true><<<g, blk, 0, stream>>>(
            v16, HID_, 0, 0, WvT, HID_, 0, 0, bv, vhT, R_, 0, 0, R_, HID_, HID_, 1);
    }

    // ---- 3. per-head qW = Q@Ww+bw, kU = K@Wu+bu (block-diagonal, f32 out) ----
    {   // per head: (768/32)*(64/32) = 48 tiles -> 12 blocks; grid.z = 8 heads
        dim3 g(12, 1, H_), blk(128);
        wmma_gemm_kernel<float, false><<<g, blk, 0, stream>>>(
            qh, HID_, 0, HD_, WwT, HD_, 0, 0, bw, qW32, HID_, 0, HD_,
            R_, HD_, HD_, H_);
        wmma_gemm_kernel<float, false><<<g, blk, 0, stream>>>(
            kh, HID_, 0, HD_, WuT, HD_, 0, 0, bu, kU32, HID_, 0, HD_,
            R_, HD_, HD_, H_);
    }

    // ---- 4. additive energy: tanh(qW+kU).vv / 8 -> attention region ----
    {
        dim3 g(L_ / 32, L_ / 32, B_ * H_), blk(16, 16);
        energy_kernel<<<g, blk, 0, stream>>>(qW32, kU32, vv, attnOut);
    }

    // ---- 5. softmax over k (in-place f32 + f16 copy) ----
    softmax_kernel<<<(B_ * H_ * L_) / 8, 256, 0, stream>>>(attnOut, attn16);

    // ---- 6. x_heads = attn @ V  (batched over (b,h), f16 out, merged heads) ----
    {   // per batch: (384/32)*(64/32) = 24 tiles -> 6 blocks; grid.z = 16
        dim3 g(6, 1, B_ * H_), blk(128);
        wmma_gemm_kernel<_Float16, false><<<g, blk, 0, stream>>>(
            attn16, L_, (long)H_ * L_ * L_, (long)L_ * L_,
            vhT, R_, (long)L_, (long)HD_ * R_,
            nullptr,
            avh, HID_, (long)L_ * HID_, (long)HD_,
            L_, HD_, L_, H_);
    }

    // ---- 7. output projection x = avh @ Wo + bo (f32 out into d_out) ----
    {
        dim3 g(96, 1, 1), blk(128);
        wmma_gemm_kernel<float, false><<<g, blk, 0, stream>>>(
            avh, HID_, 0, 0, WoT, HID_, 0, 0, bo, xOut, HID_, 0, 0,
            R_, HID_, HID_, 1);
    }

    (void)in_sizes; (void)n_in; (void)out_size; (void)ws_size;
}
